// GNNEncoderNetwork_16363825398111
// MI455X (gfx1250) — matrix operand order repaired and checked by reading, not compiled, not run on hardware
//
#include <hip/hip_runtime.h>

typedef __attribute__((ext_vector_type(16))) __bf16 v16bf;
typedef __attribute__((ext_vector_type(8)))  float  v8f;

#define D 128          // D_FEAT == D_EMB == 128
#define TILES_N 8      // 128 / 16 output-column tiles per row stripe

// ---------------------------------------------------------------------------
// Weight prep: W[k][n] f32 (row-major, in-dim first as in `state @ W`)
//   -> Wt[m][plane][n][k] bf16, plane0 = hi = bf16(w), plane1 = lo = bf16(w - hi)
// Transposed so a B fragment (one column of W, 16 consecutive K) is a
// contiguous 32-byte LDS read per lane, matching the WMMA B layout.
// ---------------------------------------------------------------------------
__global__ void prep_weights(const float* __restrict__ W, __bf16* __restrict__ Wt, int nmats)
{
    size_t total = (size_t)nmats * D * D;
    for (size_t idx = (size_t)blockIdx.x * blockDim.x + threadIdx.x; idx < total;
         idx += (size_t)gridDim.x * blockDim.x) {
        size_t m = idx / (D * D);
        size_t r = idx % (D * D);
        size_t n = r / D, k = r % D;
        float  w  = W[m * D * D + k * D + n];
        __bf16 hi = (__bf16)w;
        __bf16 lo = (__bf16)(w - (float)hi);
        Wt[m * 2 * D * D + n * D + k]         = hi;
        Wt[m * 2 * D * D + D * D + n * D + k] = lo;
    }
}

// ---------------------------------------------------------------------------
// Fused GEMM + bias + ReLU (+ optional residual add):
//   Out = [resid +] relu(A @ W + bias),  A:[nrows,128] f32, W via bf16 hi/lo.
// One workgroup = 8 waves = 128 rows. Each wave owns a 16-row stripe and all
// 8 column tiles (64 f32 accumulator VGPRs). K is consumed in 4 steps of 32
// with v_wmma_f32_16x16x32_bf16; bf16-split gives near-f32 accuracy:
//   A*B ~= Ah*Bh + Ah*Bl + Al*Bh      (Al*Bl ~ 2^-16, dropped)
// ---------------------------------------------------------------------------
__global__ void __launch_bounds__(256)
gemm_bias_relu(const float* __restrict__ A, const __bf16* __restrict__ Wt,
               const float* __restrict__ bias, const float* __restrict__ resid,
               float* __restrict__ Out, int nrows)
{
    // 64 KB LDS: 128x128 bf16 hi plane + lo plane (contiguous in Wt)
    __shared__ __align__(32) uint4 ldsraw[4096];
    const __bf16* ldsW = (const __bf16*)ldsraw;
    {
        const uint4* src = (const uint4*)Wt;
        for (int i = threadIdx.x; i < 4096; i += 256) ldsraw[i] = src[i];
    }
    __syncthreads();

    const int lane = threadIdx.x & 31;
    const int wave = threadIdx.x >> 5;
    const int half = lane >> 4;   // K-half select (A/B lane layout)
    const int m16  = lane & 15;   // row within A tile / column within B tile

    const int row0   = blockIdx.x * 128 + wave * 16;
    const int row    = row0 + m16;
    const int rclamp = row < nrows ? row : (nrows - 1);   // keep EXEC all-ones for WMMA
    const float* arow = A + (size_t)rclamp * D;

    v8f acc[TILES_N];
    const v8f zero = {0.f, 0.f, 0.f, 0.f, 0.f, 0.f, 0.f, 0.f};
#pragma unroll
    for (int t = 0; t < TILES_N; ++t) acc[t] = zero;

#pragma unroll
    for (int kt = 0; kt < 4; ++kt) {
        // A fragment, contiguous k order, same as B:
        // lane<16 : K = kt*32 + [0..15]; lane>=16: +16
        const int kb = kt * 32 + half * 16;
        float4 f0 = *(const float4*)(arow + kb);
        float4 f1 = *(const float4*)(arow + kb + 4);
        float4 f2 = *(const float4*)(arow + kb + 8);
        float4 f3 = *(const float4*)(arow + kb + 12);
        float  af[16] = {f0.x, f0.y, f0.z, f0.w, f1.x, f1.y, f1.z, f1.w,
                         f2.x, f2.y, f2.z, f2.w, f3.x, f3.y, f3.z, f3.w};
        v16bf ah, al;
#pragma unroll
        for (int j = 0; j < 16; ++j) {
            __bf16 h = (__bf16)af[j];
            ah[j] = h;
            al[j] = (__bf16)(af[j] - (float)h);
        }

        const int koff = kt * 32 + half * 16;   // B layout: lanes 16-31 hold K+16
#pragma unroll
        for (int ct = 0; ct < TILES_N; ++ct) {
            const int col = ct * 16 + m16;
            v16bf bh = *(const v16bf*)(ldsW + (size_t)col * D + koff);
            v16bf bl = *(const v16bf*)(ldsW + (size_t)(D * D) + (size_t)col * D + koff);
            acc[ct] = __builtin_amdgcn_wmma_f32_16x16x32_bf16(
                false, ah, false, bh, (short)0, acc[ct], false, false);
            acc[ct] = __builtin_amdgcn_wmma_f32_16x16x32_bf16(
                false, ah, false, bl, (short)0, acc[ct], false, false);
            acc[ct] = __builtin_amdgcn_wmma_f32_16x16x32_bf16(
                false, al, false, bh, (short)0, acc[ct], false, false);
        }
    }

    // Epilogue. C/D layout: VGPR j -> M = half*8 + j, N = lane%16.
    const bool has_resid = (resid != nullptr);
#pragma unroll
    for (int ct = 0; ct < TILES_N; ++ct) {
        const int col  = ct * 16 + m16;
        const float bc = bias[col];
#pragma unroll
        for (int j = 0; j < 8; ++j) {
            const int r = row0 + half * 8 + j;
            if (r < nrows) {
                float v = acc[ct][j] + bc;
                v = v > 0.f ? v : 0.f;
                if (has_resid) v += resid[(size_t)r * D + col];
                Out[(size_t)r * D + col] = v;
            }
        }
    }
}

// ---------------------------------------------------------------------------
// Edge scatter: agg[dst[e]] += message[src[e]]. One wave per edge, lane owns a
// float4 slice (32 lanes * 4 = 128). f32 atomics land in the 192 MB L2 (the
// 51 MB message buffer is L2-resident on MI455X).
// ---------------------------------------------------------------------------
__global__ void __launch_bounds__(256)
scatter_add(const float* __restrict__ msg, const int* __restrict__ src,
            const int* __restrict__ dst, float* __restrict__ agg, int nedges)
{
    const int lane = threadIdx.x & 31;
    const int wave = (int)((blockIdx.x * blockDim.x + threadIdx.x) >> 5);
    const int nw   = (int)((gridDim.x * blockDim.x) >> 5);
    for (int e = wave; e < nedges; e += nw) {
        const int s = src[e];
        const int d = dst[e];
        if (e + nw < nedges)   // speculative prefetch of next gather row
            __builtin_prefetch(msg + (size_t)src[e + nw] * D + lane * 4, 0, 1);
        float4 v = ((const float4*)(msg + (size_t)s * D))[lane];
        float* o = agg + (size_t)d * D + (size_t)lane * 4;
        atomicAdd(o + 0, v.x);
        atomicAdd(o + 1, v.y);
        atomicAdd(o + 2, v.z);
        atomicAdd(o + 3, v.w);
    }
}

// Sum-pool nodes into graphs: graphs[batch[n]] += state[n]
__global__ void __launch_bounds__(256)
pool_nodes(const float* __restrict__ state, const int* __restrict__ batch,
           float* __restrict__ graphs, int nnodes)
{
    const int lane = threadIdx.x & 31;
    const int wave = (int)((blockIdx.x * blockDim.x + threadIdx.x) >> 5);
    const int nw   = (int)((gridDim.x * blockDim.x) >> 5);
    for (int n = wave; n < nnodes; n += nw) {
        const int g = batch[n];
        float4 v = ((const float4*)(state + (size_t)n * D))[lane];
        float* o = graphs + (size_t)g * D + (size_t)lane * 4;
        atomicAdd(o + 0, v.x);
        atomicAdd(o + 1, v.y);
        atomicAdd(o + 2, v.z);
        atomicAdd(o + 3, v.w);
    }
}

__global__ void zero_f32(float4* __restrict__ p, long n4)
{
    long i      = (long)blockIdx.x * blockDim.x + threadIdx.x;
    long stride = (long)gridDim.x * blockDim.x;
    float4 z = {0.f, 0.f, 0.f, 0.f};
    for (; i < n4; i += stride) p[i] = z;
}

// Final projection: [G,128] @ [128,32] + b  (trivial FLOPs; plain f32 dot)
__global__ void __launch_bounds__(256)
out_proj(const float* __restrict__ G, const float* __restrict__ Wo,
         const float* __restrict__ bo, float* __restrict__ Out, int ngraphs)
{
    const int idx = blockIdx.x * blockDim.x + threadIdx.x;
    if (idx >= ngraphs * 32) return;
    const int g = idx >> 5, m = idx & 31;
    const float* gr = G + (size_t)g * D;
    float acc = bo[m];
#pragma unroll 8
    for (int k = 0; k < D; ++k) acc += gr[k] * Wo[k * 32 + m];
    Out[idx] = acc;
}

// ---------------------------------------------------------------------------
extern "C" void kernel_launch(void* const* d_in, const int* in_sizes, int n_in,
                              void* d_out, int out_size, void* d_ws, size_t ws_size,
                              hipStream_t stream)
{
    (void)n_in; (void)ws_size;
    const float* x     = (const float*)d_in[0];
    const int*   eidx  = (const int*)  d_in[1];
    const int*   batch = (const int*)  d_in[2];
    const float* emb_W = (const float*)d_in[3];
    const float* emb_b = (const float*)d_in[4];
    const float* msg_W = (const float*)d_in[5];
    const float* msg_b = (const float*)d_in[6];
    const float* upd_W = (const float*)d_in[7];
    const float* upd_b = (const float*)d_in[8];
    const float* out_W = (const float*)d_in[9];
    const float* out_b = (const float*)d_in[10];

    const int nnodes  = in_sizes[0] / D;
    const int nedges  = in_sizes[1] / 2;
    const int ngraphs = out_size / 32;
    const int* src = eidx;
    const int* dst = eidx + nedges;

    // workspace carve-up (256B aligned)
    char* ws = (char*)d_ws;
    auto carve = [&](size_t bytes) -> char* {
        char* p = ws;
        ws += (bytes + 255) & ~(size_t)255;
        return p;
    };
    float*  state   = (float*) carve((size_t)nnodes * D * sizeof(float));
    float*  message = (float*) carve((size_t)nnodes * D * sizeof(float));
    float*  agg     = (float*) carve((size_t)nnodes * D * sizeof(float));
    float*  graphs  = (float*) carve((size_t)ngraphs * D * sizeof(float));
    __bf16* embWt   = (__bf16*)carve((size_t)2 * D * D * sizeof(__bf16));
    __bf16* msgWt   = (__bf16*)carve((size_t)4 * 2 * D * D * sizeof(__bf16));
    __bf16* updWt   = (__bf16*)carve((size_t)4 * 2 * D * D * sizeof(__bf16));

    prep_weights<<<64,  256, 0, stream>>>(emb_W, embWt, 1);
    prep_weights<<<256, 256, 0, stream>>>(msg_W, msgWt, 4);
    prep_weights<<<256, 256, 0, stream>>>(upd_W, updWt, 4);

    const int  gblocks = (nnodes + 127) / 128;
    const long n4      = (long)nnodes * D / 4;

    // state = relu(x @ emb_W + emb_b)
    gemm_bias_relu<<<gblocks, 256, 0, stream>>>(x, embWt, emb_b, nullptr, state, nnodes);

    for (int r = 0; r < 4; ++r) {
        // message = relu(state @ msg_W[r] + msg_b[r])
        gemm_bias_relu<<<gblocks, 256, 0, stream>>>(
            state, msgWt + (size_t)r * 2 * D * D, msg_b + r * D, nullptr, message, nnodes);
        // aggregated = segment_sum(message[src], dst)
        zero_f32<<<2048, 256, 0, stream>>>((float4*)agg, n4);
        scatter_add<<<4096, 256, 0, stream>>>(message, src, dst, agg, nedges);
        // state = state + relu(aggregated @ upd_W[r] + upd_b[r])   (residual fused)
        gemm_bias_relu<<<gblocks, 256, 0, stream>>>(
            agg, updWt + (size_t)r * 2 * D * D, upd_b + r * D, state, state, nnodes);
    }

    // graph pooling + output projection
    zero_f32<<<64, 256, 0, stream>>>((float4*)graphs, (long)ngraphs * D / 4);
    pool_nodes<<<1024, 256, 0, stream>>>(state, batch, graphs, nnodes);
    out_proj<<<(ngraphs * 32 + 255) / 256, 256, 0, stream>>>(
        graphs, out_W, out_b, (float*)d_out, ngraphs);
}
